// MoELoRALayer_48172353192193
// MI455X (gfx1250) — matrix operand order repaired and checked by reading, not compiled, not run on hardware
//
#include <hip/hip_runtime.h>
#include <hip/hip_bf16.h>

typedef __attribute__((ext_vector_type(16))) __bf16 v16bf;
typedef __attribute__((ext_vector_type(8)))  __bf16 v8bf;
typedef __attribute__((ext_vector_type(4)))  __bf16 v4bf;
typedef __attribute__((ext_vector_type(8)))  float  v8f;

#define N_TOK 16384
#define DIN   1024
#define DOUT  1024
#define NE    8
#define RMAX  16
#define LORA_SCALE 2.0f   // ALPHA / R = 32/16

// ---------------- WMMA helpers ----------------

__device__ __forceinline__ v8f wmma_bf16(v16bf a, v16bf b, v8f c) {
  return __builtin_amdgcn_wmma_f32_16x16x32_bf16(false, a, false, b, (short)0, c,
                                                 false, false);
}

// A fragment: 16x32 bf16 tile, row-major in LDS, row stride 32 elements.
// lanes 0-15: row M=lane, K in {0..7,16..23}; lanes 16-31: M=lane-16, K in {8..15,24..31}.
__device__ __forceinline__ v16bf load_frag_a(const __bf16* lds, int lane, int row_off) {
  int r    = row_off + (lane & 15);
  int koff = (lane >> 4) << 3;
  const __bf16* p = lds + r * 32 + koff;
  v8bf lo = *(const v8bf*)(p);
  v8bf hi = *(const v8bf*)(p + 16);
  return __builtin_shufflevector(lo, hi, 0,1,2,3,4,5,6,7,8,9,10,11,12,13,14,15);
}

// B fragment: 32x16 (KxN) tile stored col-major in LDS: ldsT[col*32 + k].
__device__ __forceinline__ v16bf load_frag_b(const __bf16* ldsT, int lane, int col_off) {
  int c    = col_off + (lane & 15);
  int koff = (lane >> 4) << 3;
  const __bf16* p = ldsT + c * 32 + koff;
  v8bf lo = *(const v8bf*)(p);
  v8bf hi = *(const v8bf*)(p + 16);
  return __builtin_shufflevector(lo, hi, 0,1,2,3,4,5,6,7,8,9,10,11,12,13,14,15);
}

// Async copy of an NROWS x 32 bf16 tile (row stride gstride elems) into LDS,
// as NROWS*4 chunks of 16B via GLOBAL_LOAD_ASYNC_TO_LDS_B128 (ASYNCcnt path,
// no VGPR round trip). LDS layout == source layout (row-major, stride 32).
template <int NROWS, int NTHREADS>
__device__ __forceinline__ void async_tile(const __bf16* g0, int gstride,
                                           __bf16* l0, int tid) {
  constexpr int CHUNKS = NROWS * 4;
  constexpr int ITERS  = CHUNKS / NTHREADS;
#pragma unroll
  for (int it = 0; it < ITERS; ++it) {
    int c = tid + it * NTHREADS;
    int r = c >> 2, o = (c & 3) << 3;
    unsigned long long ga = (unsigned long long)(g0 + (size_t)r * gstride + o);
    unsigned           la = (unsigned)(unsigned long long)(l0 + r * 32 + o);
    asm volatile("global_load_async_to_lds_b128 %0, %1, off"
                 :: "v"(la), "v"(ga) : "memory");
  }
}

#define WAIT_ASYNC(n) asm volatile("s_wait_asynccnt " #n ::: "memory")

// ---------------- Kernel 1a: tokens f32 -> bf16 ----------------
__global__ void tok_bf16_kernel(const float* __restrict__ t, __bf16* __restrict__ o) {
  size_t i = ((size_t)blockIdx.x * blockDim.x + threadIdx.x) * 4;
  float4 v = *(const float4*)(t + i);
  v4bf r; r[0] = (__bf16)v.x; r[1] = (__bf16)v.y; r[2] = (__bf16)v.z; r[3] = (__bf16)v.w;
  *(v4bf*)(o + i) = r;
}

// ---------------- Kernel 1b: weight repack (transposed, bf16) ----------------
// WbT  [DOUT][DIN]  : WbT[m][k]      = Wb[k][m]
// AcatT[128][DIN]   : AcatT[e*16+r][d] = A[e][d][r]
// BcatT[DOUT][128]  : BcatT[m][e*16+r] = SCALE * B[e][r][m]
__global__ void prep_w_kernel(const float* __restrict__ Wb, const float* __restrict__ A,
                              const float* __restrict__ B, __bf16* __restrict__ WbT,
                              __bf16* __restrict__ AcatT, __bf16* __restrict__ BcatT) {
  int idx = blockIdx.x * blockDim.x + threadIdx.x;
  if (idx < DIN * DOUT) {
    int m = idx >> 10, k = idx & 1023;
    WbT[idx] = (__bf16)Wb[(size_t)k * DOUT + m];
  }
  if (idx < 128 * DIN) {
    int c = idx >> 10, d = idx & 1023;
    int e = c >> 4, r = c & 15;
    AcatT[idx] = (__bf16)A[((size_t)e * DIN + d) * RMAX + r];
  }
  if (idx < DOUT * 128) {
    int m = idx >> 7, c = idx & 127;
    BcatT[idx] = (__bf16)(LORA_SCALE * B[(size_t)c * DOUT + m]);
  }
}

// ---------------- Kernel 2: router (one wave per token) ----------------
__global__ void router_kernel(const float* __restrict__ tokens,
                              const float* __restrict__ Wg,
                              float* __restrict__ logits_out,
                              float* __restrict__ sel_out,
                              float* __restrict__ wts_out,
                              float* __restrict__ gates) {
  int gwave = (blockIdx.x * blockDim.x + threadIdx.x) >> 5;
  int lane  = threadIdx.x & 31;
  const float* trow = tokens + (size_t)gwave * DIN;

  float acc[NE];
#pragma unroll
  for (int e = 0; e < NE; ++e) acc[e] = 0.f;

  for (int d = lane; d < DIN; d += 32) {
    float t = trow[d];
    const float* g = Wg + (size_t)d * NE;
#pragma unroll
    for (int e = 0; e < NE; ++e) acc[e] += t * g[e];
  }
#pragma unroll
  for (int e = 0; e < NE; ++e)
    for (int off = 16; off > 0; off >>= 1)
      acc[e] += __shfl_xor(acc[e], off, 32);

  if (lane == 0) {
    int   i0 = 0;
    float v0 = acc[0];
#pragma unroll
    for (int e = 1; e < NE; ++e)
      if (acc[e] > v0) { v0 = acc[e]; i0 = e; }
    int   i1 = -1;
    float v1 = -3.4e38f;
#pragma unroll
    for (int e = 0; e < NE; ++e)
      if (e != i0 && acc[e] > v1) { v1 = acc[e]; i1 = e; }

    float e1  = __expf(v1 - v0);
    float inv = 1.f / (1.f + e1);
    float w0 = inv, w1 = e1 * inv;

    size_t n = (size_t)gwave;
#pragma unroll
    for (int e = 0; e < NE; ++e) logits_out[n * NE + e] = acc[e];
    sel_out[n * 2 + 0] = (float)i0;
    sel_out[n * 2 + 1] = (float)i1;
    wts_out[n * 2 + 0] = w0;
    wts_out[n * 2 + 1] = w1;
#pragma unroll
    for (int e = 0; e < NE; ++e)
      gates[n * NE + e] = (e == i0) ? w0 : ((e == i1) ? w1 : 0.f);
  }
}

// ---------------- Kernel 3: H = gate .* (tokens @ Acat), bf16 ----------------
// block: 128 threads (4 waves); tile 64 rows x 128 cols; col tile t == expert t.
// Double-buffered async global->LDS; 6 async instructions per wave per tile.
__global__ void lora_h_kernel(const __bf16* __restrict__ tok,
                              const __bf16* __restrict__ AcatT,
                              const float* __restrict__ gates,
                              __bf16* __restrict__ H) {
  __shared__ __align__(16) __bf16 sA[2][64 * 32];
  __shared__ __align__(16) __bf16 sB[2][128 * 32];

  int tid  = threadIdx.x;
  int wave = tid >> 5;
  int lane = tid & 31;
  int row0 = blockIdx.x * 64;

  const __bf16* Abase = tok + (size_t)row0 * DIN;
  const __bf16* Bbase = AcatT;   // 128 rows x DIN

  v8f acc[8];
#pragma unroll
  for (int t = 0; t < 8; ++t) acc[t] = (v8f)0.f;

  async_tile<64, 128>(Abase, DIN, sA[0], tid);
  async_tile<128, 128>(Bbase, DIN, sB[0], tid);

  const int nsteps = DIN / 32;
  for (int s = 0; s < nsteps; ++s) {
    int cur = s & 1;
    if (s + 1 < nsteps) {
      async_tile<64, 128>(Abase + (s + 1) * 32, DIN, sA[cur ^ 1], tid);
      async_tile<128, 128>(Bbase + (s + 1) * 32, DIN, sB[cur ^ 1], tid);
      WAIT_ASYNC(0x6);
    } else {
      WAIT_ASYNC(0x0);
    }
    __syncthreads();

    v16bf a = load_frag_a(sA[cur], lane, wave * 16);
#pragma unroll
    for (int t = 0; t < 8; ++t)
      acc[t] = wmma_bf16(a, load_frag_b(sB[cur], lane, t * 16), acc[t]);
    __syncthreads();
  }

  int col   = lane & 15;
  int rbase = row0 + wave * 16 + ((lane >> 4) << 3);
#pragma unroll
  for (int t = 0; t < 8; ++t) {
#pragma unroll
    for (int i = 0; i < 8; ++i) {
      int row = rbase + i;
      float g = gates[(size_t)row * NE + t];
      H[(size_t)row * 128 + t * 16 + col] = (__bf16)(acc[t][i] * g);
    }
  }
}

// ---------------- Kernel 4: out = tokens@W_base + H@Bcat ----------------
// block: 256 threads (8 waves); tile 128 rows x 64 cols; double-buffered async.
// 3 async instructions per wave per tile.
__global__ void moe_out_kernel(const __bf16* __restrict__ tok,
                               const __bf16* __restrict__ WbT,
                               const __bf16* __restrict__ H,
                               const __bf16* __restrict__ BcatT,
                               float* __restrict__ out) {
  __shared__ __align__(16) __bf16 sA[2][128 * 32];
  __shared__ __align__(16) __bf16 sB[2][64 * 32];

  int tid  = threadIdx.x;
  int wave = tid >> 5;
  int lane = tid & 31;
  int row0 = blockIdx.x * 128;
  int col0 = blockIdx.y * 64;

  v8f acc[4];
#pragma unroll
  for (int t = 0; t < 4; ++t) acc[t] = (v8f)0.f;

  // Phase 1: K over D_IN (tokens @ W_base)
  {
    const __bf16* Abase = tok + (size_t)row0 * DIN;
    const __bf16* Bbase = WbT + (size_t)col0 * DIN;
    async_tile<128, 256>(Abase, DIN, sA[0], tid);
    async_tile<64, 256>(Bbase, DIN, sB[0], tid);
    const int nsteps = DIN / 32;
    for (int s = 0; s < nsteps; ++s) {
      int cur = s & 1;
      if (s + 1 < nsteps) {
        async_tile<128, 256>(Abase + (s + 1) * 32, DIN, sA[cur ^ 1], tid);
        async_tile<64, 256>(Bbase + (s + 1) * 32, DIN, sB[cur ^ 1], tid);
        WAIT_ASYNC(0x3);
      } else {
        WAIT_ASYNC(0x0);
      }
      __syncthreads();
      v16bf a = load_frag_a(sA[cur], lane, wave * 16);
#pragma unroll
      for (int t = 0; t < 4; ++t)
        acc[t] = wmma_bf16(a, load_frag_b(sB[cur], lane, t * 16), acc[t]);
      __syncthreads();
    }
  }

  // Phase 2: K over 128 (H @ Bcat), same accumulators
  {
    const __bf16* Abase = H + (size_t)row0 * 128;
    const __bf16* Bbase = BcatT + (size_t)col0 * 128;
    async_tile<128, 256>(Abase, 128, sA[0], tid);
    async_tile<64, 256>(Bbase, 128, sB[0], tid);
    const int nsteps = 128 / 32;
    for (int s = 0; s < nsteps; ++s) {
      int cur = s & 1;
      if (s + 1 < nsteps) {
        async_tile<128, 256>(Abase + (s + 1) * 32, 128, sA[cur ^ 1], tid);
        async_tile<64, 256>(Bbase + (s + 1) * 32, 128, sB[cur ^ 1], tid);
        WAIT_ASYNC(0x3);
      } else {
        WAIT_ASYNC(0x0);
      }
      __syncthreads();
      v16bf a = load_frag_a(sA[cur], lane, wave * 16);
#pragma unroll
      for (int t = 0; t < 4; ++t)
        acc[t] = wmma_bf16(a, load_frag_b(sB[cur], lane, t * 16), acc[t]);
      __syncthreads();
    }
  }

  int col   = lane & 15;
  int rbase = row0 + wave * 16 + ((lane >> 4) << 3);
#pragma unroll
  for (int t = 0; t < 4; ++t) {
#pragma unroll
    for (int i = 0; i < 8; ++i)
      out[(size_t)(rbase + i) * DOUT + col0 + t * 16 + col] = acc[t][i];
  }
}

// ---------------- Launch ----------------

extern "C" void kernel_launch(void* const* d_in, const int* in_sizes, int n_in,
                              void* d_out, int out_size, void* d_ws, size_t ws_size,
                              hipStream_t stream) {
  const float* tokens = (const float*)d_in[0];
  const float* W_base = (const float*)d_in[1];
  const float* A      = (const float*)d_in[2];
  const float* B      = (const float*)d_in[3];
  const float* W_gate = (const float*)d_in[4];

  float* out    = (float*)d_out;                      // [N, D_OUT]
  float* logits = out + (size_t)N_TOK * DOUT;         // [N, E]
  float* sel    = logits + (size_t)N_TOK * NE;        // [N, 2] (indices as floats)
  float* wts    = sel + (size_t)N_TOK * 2;            // [N, 2]

  char*   ws     = (char*)d_ws;
  float*  gates  = (float*)ws;                        // 512 KB
  __bf16* WbT    = (__bf16*)(ws + (size_t)(512 << 10));           // 2 MB
  __bf16* AcatT  = (__bf16*)(ws + (size_t)(2560 << 10));          // 256 KB
  __bf16* BcatT  = (__bf16*)(ws + (size_t)(2816 << 10));          // 256 KB
  __bf16* H      = (__bf16*)(ws + (size_t)(3072 << 10));          // 4 MB
  __bf16* tokb   = (__bf16*)(ws + (size_t)(7168 << 10));          // 32 MB

  tok_bf16_kernel<<<(N_TOK * DIN / 4) / 256, 256, 0, stream>>>(tokens, tokb);
  prep_w_kernel<<<(DIN * DOUT) / 256, 256, 0, stream>>>(W_base, A, B, WbT, AcatT, BcatT);
  router_kernel<<<N_TOK / 8, 256, 0, stream>>>(tokens, W_gate, logits, sel, wts, gates);
  lora_h_kernel<<<N_TOK / 64, 128, 0, stream>>>(tokb, AcatT, gates, H);
  moe_out_kernel<<<dim3(N_TOK / 128, DOUT / 64), 256, 0, stream>>>(tokb, WbT, H,
                                                                   BcatT, out);
}
